// TargetNet_16011638079542
// MI455X (gfx1250) — matrix operand order repaired and checked by reading, not compiled, not run on hardware
//
#include <hip/hip_runtime.h>

// ---------------------------------------------------------------------------
// B=64 per-sample MLPs over 256 pixels: 2048->1024->512->256->128->1, sigmoid.
// Strategy: activations kept pixel-major [B][256][C] in f16; fp32 weights
// converted to f16 in-register; all MACs via v_wmma_f32_16x16x32_f16 with
// f32 accumulation. ~91 GFLOP over ~0.84 GB read -> near HBM-bound with the
// f16 WMMA pipe on MI455X.
// Workspace usage: ping-pong buffers at d_ws+0 (64MB) and d_ws+64MB (32MB).
// ---------------------------------------------------------------------------

typedef __attribute__((ext_vector_type(16))) _Float16 v16h;
typedef __attribute__((ext_vector_type(8)))  _Float16 v8h;
typedef __attribute__((ext_vector_type(8)))  float    v8f;

#define NPIX 256   // 16x16 pixels
#define NB   64    // batch

// ---------------------------------------------------------------------------
// Transpose + convert: x [B][C][256] fp32 -> XT [B][256][C] f16
// LDS-tiled 32x32 so both global read and write are fully coalesced.
// ---------------------------------------------------------------------------
__global__ __launch_bounds__(256)
void transpose_cvt_kernel(const float* __restrict__ x, _Float16* __restrict__ xt, int C) {
    __shared__ _Float16 tile[32][34];           // 34 halves = 68B row stride, co-prime with banks
    const int b  = blockIdx.z;
    const int c0 = blockIdx.y * 32;
    const int p0 = blockIdx.x * 32;
    const int tx = threadIdx.x;                  // 0..31
    #pragma unroll
    for (int i = threadIdx.y; i < 32; i += 8)
        tile[i][tx] = (_Float16)x[((size_t)b * C + c0 + i) * NPIX + p0 + tx];
    __syncthreads();
    #pragma unroll
    for (int i = threadIdx.y; i < 32; i += 8)
        xt[((size_t)b * NPIX + p0 + i) * (size_t)C + c0 + tx] = tile[tx][i];
}

// ---------------------------------------------------------------------------
// Sigmoid + pack one 16x16 f32 accumulator tile to f16, contiguous v8h store.
// Lane l<16: rows M=m0+0..7, col N=n+l ; lane>=16: rows M=m0+8..15, col N=n+l-16
// -> pixel-major store ptr[lane row][m0 + hi*8 .. +7] is 16B contiguous.
// ---------------------------------------------------------------------------
__device__ __forceinline__
void store_tile_sigmoid(v8f acc, const float* bvs, _Float16* ptr) {
    v8h o;
    #pragma unroll
    for (int r = 0; r < 8; ++r) {
        float v = acc[r] + bvs[r];
        v = 1.0f / (1.0f + __expf(-v));
        o[r] = (_Float16)v;
    }
    *(v8h*)ptr = o;
}

// ---------------------------------------------------------------------------
// One layer: Hout[b][n][m] = sigmoid( sum_k W[b][m][k] * Hin[b][n][k] + bias[b][m] )
// Hin:  [B][256][K] f16 (pixel-major)    W: [B][M][K] fp32    bias: [B][M] fp32
// Hout: [B][256][M] f16 (pixel-major)
// Block = 128 threads = 4 waves; wave w owns pixels [64w, 64w+64) for one
// 16-row M tile: 4 WMMA accumulators, A fragment loaded once per k-step.
// ---------------------------------------------------------------------------
__global__ __launch_bounds__(128)
void fc_sigmoid_kernel(const _Float16* __restrict__ Hin,
                       const float*    __restrict__ W,
                       const float*    __restrict__ bias,
                       _Float16*       __restrict__ Hout,
                       int K, int M, int Mtiles)
{
    const int wave = threadIdx.x >> 5;          // 0..3
    const int lane = threadIdx.x & 31;
    const int ln   = lane & 15;
    const int hi   = lane >> 4;                 // half-wave select
    const int mt   = blockIdx.x % Mtiles;
    const int bb   = blockIdx.x / Mtiles;
    const int m0   = mt * 16;
    const int n0   = wave * 64;

    // A: lane<16 -> row m0+ln, K k0+0..15; lane>=16 -> K k0+16..31
    const float*    wrow  = W   + ((size_t)bb * M + m0 + ln) * (size_t)K + hi * 16;
    // B: lane<16 -> pixel n0+ln, K k0+0..15; lane>=16 -> K k0+16..31
    const _Float16* hbase = Hin + ((size_t)bb * NPIX + n0 + ln) * (size_t)K + hi * 16;
    const size_t rowK = (size_t)16 * K;         // 16 pixels ahead

    v8f acc0 = {}, acc1 = {}, acc2 = {}, acc3 = {};

    for (int k0 = 0; k0 < K; k0 += 32) {
        // ---- A fragment: 4x float4 fp32 loads, convert to f16 in-register
        float4 f0 = *(const float4*)(wrow + k0);
        float4 f1 = *(const float4*)(wrow + k0 + 4);
        float4 f2 = *(const float4*)(wrow + k0 + 8);
        float4 f3 = *(const float4*)(wrow + k0 + 12);
        v16h a;
        a[0]=(_Float16)f0.x;  a[1]=(_Float16)f0.y;  a[2]=(_Float16)f0.z;  a[3]=(_Float16)f0.w;
        a[4]=(_Float16)f1.x;  a[5]=(_Float16)f1.y;  a[6]=(_Float16)f1.z;  a[7]=(_Float16)f1.w;
        a[8]=(_Float16)f2.x;  a[9]=(_Float16)f2.y;  a[10]=(_Float16)f2.z; a[11]=(_Float16)f2.w;
        a[12]=(_Float16)f3.x; a[13]=(_Float16)f3.y; a[14]=(_Float16)f3.z; a[15]=(_Float16)f3.w;

        // ---- B fragments: contiguous 32B per lane, 4 pixel tiles
        v16h b0 = *(const v16h*)(hbase + k0);
        v16h b1 = *(const v16h*)(hbase + rowK     + k0);
        v16h b2 = *(const v16h*)(hbase + rowK * 2 + k0);
        v16h b3 = *(const v16h*)(hbase + rowK * 3 + k0);

        acc0 = __builtin_amdgcn_wmma_f32_16x16x32_f16(false, a, false, b0, (short)0, acc0, false, false);
        acc1 = __builtin_amdgcn_wmma_f32_16x16x32_f16(false, a, false, b1, (short)0, acc1, false, false);
        acc2 = __builtin_amdgcn_wmma_f32_16x16x32_f16(false, a, false, b2, (short)0, acc2, false, false);
        acc3 = __builtin_amdgcn_wmma_f32_16x16x32_f16(false, a, false, b3, (short)0, acc3, false, false);
    }

    // Bias for rows m0 + hi*8 .. +7 (contiguous)
    const float* bp = bias + (size_t)bb * M + m0 + hi * 8;
    float4 bv0 = *(const float4*)bp;
    float4 bv1 = *(const float4*)(bp + 4);
    float bvs[8] = {bv0.x, bv0.y, bv0.z, bv0.w, bv1.x, bv1.y, bv1.z, bv1.w};

    _Float16* obase = Hout + ((size_t)bb * NPIX + n0 + ln) * (size_t)M + m0 + hi * 8;
    const size_t rowM = (size_t)16 * M;
    store_tile_sigmoid(acc0, bvs, obase);
    store_tile_sigmoid(acc1, bvs, obase + rowM);
    store_tile_sigmoid(acc2, bvs, obase + rowM * 2);
    store_tile_sigmoid(acc3, bvs, obase + rowM * 3);
}

// ---------------------------------------------------------------------------
// fc5: out[b][p] = sigmoid( dot(W5[b][0][0..127], H4[b][p][0..127]) + b5[b] )
// Tiny (~0.005% of FLOPs) -> simple VALU kernel, one thread per output.
// ---------------------------------------------------------------------------
__global__ __launch_bounds__(256)
void fc5_kernel(const _Float16* __restrict__ H4,   // [B][256][128]
                const float*    __restrict__ W5,   // [B][128]
                const float*    __restrict__ B5,   // [B]
                float*          __restrict__ out)  // [B*256]
{
    const int idx = blockIdx.x * blockDim.x + threadIdx.x;  // 0..16383
    const int bb  = idx >> 8;
    const _Float16* h = H4 + (size_t)idx * 128;
    const float*    w = W5 + (size_t)bb * 128;
    float acc = 0.0f;
    #pragma unroll
    for (int i = 0; i < 128; i += 8) {
        v8h hv = *(const v8h*)(h + i);
        float4 w0 = *(const float4*)(w + i);
        float4 w1 = *(const float4*)(w + i + 4);
        acc += (float)hv[0] * w0.x + (float)hv[1] * w0.y +
               (float)hv[2] * w0.z + (float)hv[3] * w0.w +
               (float)hv[4] * w1.x + (float)hv[5] * w1.y +
               (float)hv[6] * w1.z + (float)hv[7] * w1.w;
    }
    acc += B5[bb];
    out[idx] = 1.0f / (1.0f + __expf(-acc));
}

// ---------------------------------------------------------------------------
extern "C" void kernel_launch(void* const* d_in, const int* in_sizes, int n_in,
                              void* d_out, int out_size, void* d_ws, size_t ws_size,
                              hipStream_t stream) {
    const float* x  = (const float*)d_in[0];
    const float* w1 = (const float*)d_in[1];  const float* b1 = (const float*)d_in[2];
    const float* w2 = (const float*)d_in[3];  const float* b2 = (const float*)d_in[4];
    const float* w3 = (const float*)d_in[5];  const float* b3 = (const float*)d_in[6];
    const float* w4 = (const float*)d_in[7];  const float* b4 = (const float*)d_in[8];
    const float* w5 = (const float*)d_in[9];  const float* b5 = (const float*)d_in[10];

    // Ping-pong workspace: buf0 (64MB: XT / H2 / H4), buf1 (32MB: H1 / H3)
    char* ws = (char*)d_ws;
    _Float16* buf0 = (_Float16*)ws;                          // max 64*256*2048*2 = 64MB
    _Float16* buf1 = (_Float16*)(ws + ((size_t)64 << 20));   // max 64*256*1024*2 = 32MB
    _Float16* XT = buf0;   // [B][256][2048]
    _Float16* H1 = buf1;   // [B][256][1024]
    _Float16* H2 = buf0;   // [B][256][512]   (XT no longer needed)
    _Float16* H3 = buf1;   // [B][256][256]   (H1 no longer needed)
    _Float16* H4 = buf0;   // [B][256][128]   (H2 no longer needed)

    transpose_cvt_kernel<<<dim3(8, 64, NB), dim3(32, 8), 0, stream>>>(x, XT, 2048);

    // grid = B * Mtiles blocks, 128 threads (4 waves, 64 pixels each)
    fc_sigmoid_kernel<<<NB * 64, 128, 0, stream>>>(XT, w1, b1, H1, 2048, 1024, 64);
    fc_sigmoid_kernel<<<NB * 32, 128, 0, stream>>>(H1, w2, b2, H2, 1024,  512, 32);
    fc_sigmoid_kernel<<<NB * 16, 128, 0, stream>>>(H2, w3, b3, H3,  512,  256, 16);
    fc_sigmoid_kernel<<<NB *  8, 128, 0, stream>>>(H3, w4, b4, H4,  256,  128,  8);

    fc5_kernel<<<NB, 256, 0, stream>>>(H4, w5, b5, (float*)d_out);
}
